// Transformer_7344394076780
// MI455X (gfx1250) — matrix operand order repaired and checked by reading, not compiled, not run on hardware
//
#include <hip/hip_runtime.h>

#define B_   2
#define S_   2048
#define D_   256
#define H_   8
#define DK_  32
#define DV_  32
#define DFF_ 512
#define NL_  2
#define EPS2_ 1.42108547152020037e-14f

typedef __bf16 bf16;
typedef __attribute__((ext_vector_type(16))) __bf16 v16bf;
typedef __attribute__((ext_vector_type(8)))  __bf16 v8bf;
typedef __attribute__((ext_vector_type(8)))  float  v8f;

union V16U { v16bf v; v8bf h[2]; };

__device__ __forceinline__ bf16 f2bf(float f) { return (bf16)f; }

// ---- WMMA operand loaders (CDNA5 16-bit layouts, wave32) -------------------
// A (16x32, MxK): lane = {m = lane&15, hf = lane>>4};
//   elems 0..7  -> K = 8*hf + e            (contiguous)
//   elems 8..15 -> K = 16 + 8*hf + (e-8)   (contiguous)
__device__ __forceinline__ v16bf load_a16x32(const bf16* base, int lda, int lane) {
  int m = lane & 15, hf = lane >> 4;
  V16U u;
  u.h[0] = *(const v8bf*)(base + (size_t)m * lda + hf * 8);
  u.h[1] = *(const v8bf*)(base + (size_t)m * lda + 16 + hf * 8);
  return u.v;
}
// B (32x16, KxN) from N-major (transposed) storage: row n of baseT is the
// K-contiguous column n of B. lane = {n = lane&15, hf = lane>>4};
//   elems 0..7 -> K = 8*hf + e; elems 8..15 -> K = 16 + 8*hf + (e-8)
__device__ __forceinline__ v16bf load_b32x16(const bf16* baseT, int ldb, int lane) {
  int n = lane & 15, hf = lane >> 4;
  V16U u;
  u.h[0] = *(const v8bf*)(baseT + (size_t)n * ldb + hf * 8);
  u.h[1] = *(const v8bf*)(baseT + (size_t)n * ldb + 16 + hf * 8);
  return u.v;
}

__device__ __forceinline__ v8f wmma_bf16(v16bf a, v16bf b, v8f c) {
  return __builtin_amdgcn_wmma_f32_16x16x32_bf16(false, a, false, b, (short)0, c,
                                                 false, false);
}

// ---- 16-lane butterfly reductions via ds_swizzle (immediate pattern) --------
// group-of-32 mode: offset = {and=0x1F}<<10 | {or=0}<<5 | xor_mask
template<int IMM>
__device__ __forceinline__ float swz(float x) {
  return __int_as_float(__builtin_amdgcn_ds_swizzle(__float_as_int(x), IMM));
}
__device__ __forceinline__ float red_max16(float v) {
  v = fmaxf(v, swz<0x7C01>(v));
  v = fmaxf(v, swz<0x7C02>(v));
  v = fmaxf(v, swz<0x7C04>(v));
  v = fmaxf(v, swz<0x7C08>(v));
  return v;
}
__device__ __forceinline__ float red_sum16(float v) {
  v += swz<0x7C01>(v);
  v += swz<0x7C02>(v);
  v += swz<0x7C04>(v);
  v += swz<0x7C08>(v);
  return v;
}

// ---- weight transpose + f32->bf16 ------------------------------------------
__global__ void transpose_cvt_kernel(const float* __restrict__ in, bf16* __restrict__ out,
                                     int rows, int cols) {
  int c = blockIdx.x * blockDim.x + threadIdx.x;
  int r = blockIdx.y * blockDim.y + threadIdx.y;
  size_t off = (size_t)blockIdx.z * rows * cols;
  if (r < rows && c < cols)
    out[off + (size_t)c * rows + r] = f2bf(in[off + (size_t)r * cols + c]);
}

__global__ void cvt_copy_kernel(const float* __restrict__ in, float* __restrict__ outF,
                                bf16* __restrict__ outB, int n) {
  int i = blockIdx.x * blockDim.x + threadIdx.x;
  if (i < n) { float v = in[i]; outF[i] = v; outB[i] = f2bf(v); }
}

// ---- generic WMMA GEMM, 32x32 register tile per wave ------------------------
// C[M,N] = A[M,K](bf16,row-major) x B (via Bt[N,K], K-contiguous columns)
template<bool HAS_BIAS, bool RELU, bool HAS_RES, bool OUT_F32, bool OUT_BF16>
__global__ __launch_bounds__(32)
void gemm_bf16_kernel(const bf16* __restrict__ A, const bf16* __restrict__ Bt,
                      const float* __restrict__ bias, const float* __restrict__ res,
                      float* __restrict__ outF, bf16* __restrict__ outB,
                      int M, int N, int K) {
  int lane = threadIdx.x & 31;
  int m0 = blockIdx.x * 32;
  int n0 = blockIdx.y * 32;
  const bf16* a = A  + (size_t)m0 * K;
  const bf16* b = Bt + (size_t)n0 * K;
  v8f acc00 = {}, acc01 = {}, acc10 = {}, acc11 = {};
  for (int k = 0; k < K; k += 32) {
    if (k + 32 < K) {
      __builtin_prefetch(a + k + 32, 0, 1);
      __builtin_prefetch(b + k + 32, 0, 1);
    }
    v16bf a0 = load_a16x32(a + k, K, lane);
    v16bf a1 = load_a16x32(a + (size_t)16 * K + k, K, lane);
    v16bf b0 = load_b32x16(b + k, K, lane);
    v16bf b1 = load_b32x16(b + (size_t)16 * K + k, K, lane);
    acc00 = wmma_bf16(a0, b0, acc00);
    acc01 = wmma_bf16(a0, b1, acc01);
    acc10 = wmma_bf16(a1, b0, acc10);
    acc11 = wmma_bf16(a1, b1, acc11);
  }
  int n = lane & 15, hf = lane >> 4;
  v8f* accs[2][2] = {{&acc00, &acc01}, {&acc10, &acc11}};
#pragma unroll
  for (int i = 0; i < 2; ++i) {
#pragma unroll
    for (int j = 0; j < 2; ++j) {
      int col = n0 + 16 * j + n;
      float bia = HAS_BIAS ? bias[col] : 0.0f;
      v8f acc = *accs[i][j];
#pragma unroll
      for (int r = 0; r < 8; ++r) {
        int row = m0 + 16 * i + r + 8 * hf;
        float v = acc[r] + bia;
        if (HAS_RES) v += res[(size_t)row * N + col];
        if (RELU)    v = v > 0.0f ? v : 0.0f;
        if (OUT_F32)  outF[(size_t)row * N + col] = v;
        if (OUT_BF16) outB[(size_t)row * N + col] = f2bf(v);
      }
    }
  }
}

// ---- QKV projection, 32 rows x DK(=32) per wave -----------------------------
// TRANS_OUT=false: out[bh][S][DK] row-major (Q,K). true: out[bh][DK][S] (V^T).
template<bool TRANS_OUT>
__global__ __launch_bounds__(32)
void qkv_gemm_kernel(const bf16* __restrict__ xb, const bf16* __restrict__ Wt,
                     bf16* __restrict__ out) {
  int lane = threadIdx.x & 31;
  int m0 = blockIdx.x * 32;
  int bh = blockIdx.z;
  int b = bh / H_, h = bh % H_;
  const bf16* A  = xb + (size_t)b * S_ * D_ + (size_t)m0 * D_;
  const bf16* Bt = Wt + (size_t)h * DK_ * D_;
  v8f acc00 = {}, acc01 = {}, acc10 = {}, acc11 = {};
  for (int k = 0; k < D_; k += 32) {
    if (k + 32 < D_) {
      __builtin_prefetch(A + k + 32, 0, 1);
    }
    v16bf a0 = load_a16x32(A + k, D_, lane);
    v16bf a1 = load_a16x32(A + (size_t)16 * D_ + k, D_, lane);
    v16bf b0 = load_b32x16(Bt + k, D_, lane);
    v16bf b1 = load_b32x16(Bt + (size_t)16 * D_ + k, D_, lane);
    acc00 = wmma_bf16(a0, b0, acc00);
    acc01 = wmma_bf16(a0, b1, acc01);
    acc10 = wmma_bf16(a1, b0, acc10);
    acc11 = wmma_bf16(a1, b1, acc11);
  }
  int n = lane & 15, hf = lane >> 4;
  v8f* accs[2][2] = {{&acc00, &acc01}, {&acc10, &acc11}};
#pragma unroll
  for (int i = 0; i < 2; ++i) {
#pragma unroll
    for (int j = 0; j < 2; ++j) {
      v8f acc = *accs[i][j];
      if (TRANS_OUT) {
        bf16 tmp[8];
#pragma unroll
        for (int r = 0; r < 8; ++r) tmp[r] = f2bf(acc[r]);
        *(v8bf*)(out + ((size_t)bh * DK_ + 16 * j + n) * S_ + m0 + 16 * i + 8 * hf) =
            *(v8bf*)tmp;
      } else {
#pragma unroll
        for (int r = 0; r < 8; ++r)
          out[((size_t)bh * S_ + m0 + 16 * i + r + 8 * hf) * DK_ + 16 * j + n] =
              f2bf(acc[r]);
      }
    }
  }
}

// ---- flash attention (no 1/sqrt(dk) scale, per reference) -------------------
// grid: (S/32, B*H); one wave per 32-row query tile; K/V tiles reused across
// both 16-row sub-tiles.
__global__ __launch_bounds__(32)
void flash_attn_kernel(const bf16* __restrict__ Q, const bf16* __restrict__ Km,
                       const bf16* __restrict__ Vt, bf16* __restrict__ conc) {
  int lane = threadIdx.x & 31;
  int q0 = blockIdx.x * 32;
  int bh = blockIdx.y;
  int b = bh / H_, h = bh % H_;
  const bf16* Qbh = Q  + (size_t)bh * S_ * DK_;
  const bf16* Kbh = Km + (size_t)bh * S_ * DK_;
  const bf16* Vbh = Vt + (size_t)bh * DV_ * S_;   // [DV][S]

  __shared__ __align__(16) bf16 plds[2][16 * 32];

  v16bf qa0 = load_a16x32(Qbh + (size_t)q0 * DK_, DK_, lane);
  v16bf qa1 = load_a16x32(Qbh + (size_t)(q0 + 16) * DK_, DK_, lane);
  int n = lane & 15, hf = lane >> 4;

  float mrow[2][8], lrow[2][8];
#pragma unroll
  for (int i = 0; i < 2; ++i)
#pragma unroll
    for (int r = 0; r < 8; ++r) { mrow[i][r] = -3.0e38f; lrow[i][r] = 0.0f; }
  v8f O00 = {}, O01 = {}, O10 = {}, O11 = {};

  for (int s = 0; s < S_; s += 32) {
    if (s + 32 < S_) {
      __builtin_prefetch(Kbh + (size_t)(s + 32) * DK_, 0, 1);
      __builtin_prefetch(Vbh + s + 32, 0, 1);
    }
    v16bf kb0 = load_b32x16(Kbh + (size_t)s * DK_, DK_, lane);
    v16bf kb1 = load_b32x16(Kbh + (size_t)(s + 16) * DK_, DK_, lane);
    v8f st[2][2];
    st[0][0] = wmma_bf16(qa0, kb0, v8f{});
    st[0][1] = wmma_bf16(qa0, kb1, v8f{});
    st[1][0] = wmma_bf16(qa1, kb0, v8f{});
    st[1][1] = wmma_bf16(qa1, kb1, v8f{});

    __syncthreads();
#pragma unroll
    for (int i = 0; i < 2; ++i) {
      float p0[8], p1[8];
#pragma unroll
      for (int r = 0; r < 8; ++r) {
        float rmax = red_max16(fmaxf(st[i][0][r], st[i][1][r]));
        float mn = fmaxf(mrow[i][r], rmax);
        float scale = __expf(mrow[i][r] - mn);
        mrow[i][r] = mn;
        p0[r] = __expf(st[i][0][r] - mn);
        p1[r] = __expf(st[i][1][r] - mn);
        float rsum = red_sum16(p0[r] + p1[r]);
        lrow[i][r] = lrow[i][r] * scale + rsum;
        if (i == 0) { O00[r] *= scale; O01[r] *= scale; }
        else        { O10[r] *= scale; O11[r] *= scale; }
      }
      // C-layout -> A-layout via LDS (16x32 bf16 probabilities)
#pragma unroll
      for (int r = 0; r < 8; ++r) {
        int row = r + 8 * hf;
        plds[i][row * 32 + n]      = f2bf(p0[r]);
        plds[i][row * 32 + 16 + n] = f2bf(p1[r]);
      }
    }
    __syncthreads();
    v16bf pa0 = load_a16x32(plds[0], 32, lane);
    v16bf pa1 = load_a16x32(plds[1], 32, lane);

    v16bf vb0 = load_b32x16(Vbh + s, S_, lane);                   // dv 0..15
    v16bf vb1 = load_b32x16(Vbh + (size_t)16 * S_ + s, S_, lane); // dv 16..31
    O00 = wmma_bf16(pa0, vb0, O00);
    O01 = wmma_bf16(pa0, vb1, O01);
    O10 = wmma_bf16(pa1, vb0, O10);
    O11 = wmma_bf16(pa1, vb1, O11);
  }

  v8f* Os[2][2] = {{&O00, &O01}, {&O10, &O11}};
#pragma unroll
  for (int i = 0; i < 2; ++i)
#pragma unroll
    for (int j = 0; j < 2; ++j) {
      v8f O = *Os[i][j];
#pragma unroll
      for (int r = 0; r < 8; ++r) {
        int row = q0 + 16 * i + r + 8 * hf;
        float inv = 1.0f / lrow[i][r];
        conc[((size_t)b * S_ + row) * (H_ * DV_) + (size_t)h * DV_ + 16 * j + n] =
            f2bf(O[r] * inv);
      }
    }
}

// ---- layernorm: scalar gamma/beta; writes f32 residual + bf16 activation ----
__global__ __launch_bounds__(256)
void layernorm_kernel(const float* __restrict__ in, const float* __restrict__ gamma,
                      const float* __restrict__ beta, float* __restrict__ outF,
                      bf16* __restrict__ outB) {
  int row = blockIdx.x, tid = threadIdx.x;
  __shared__ float red[256];
  float v = in[(size_t)row * D_ + tid];
  red[tid] = v;
  __syncthreads();
  for (int s = 128; s > 0; s >>= 1) {
    if (tid < s) red[tid] += red[tid + s];
    __syncthreads();
  }
  float mean = red[0] * (1.0f / D_);
  __syncthreads();
  float d = v - mean;
  red[tid] = d * d;
  __syncthreads();
  for (int s = 128; s > 0; s >>= 1) {
    if (tid < s) red[tid] += red[tid + s];
    __syncthreads();
  }
  float var = red[0] * (1.0f / D_);
  float y = d * rsqrtf(var + EPS2_) * gamma[0] + beta[0];
  outF[(size_t)row * D_ + tid] = y;
  outB[(size_t)row * D_ + tid] = f2bf(y);
}

// ---- host launcher ----------------------------------------------------------
extern "C" void kernel_launch(void* const* d_in, const int* in_sizes, int n_in,
                              void* d_out, int out_size, void* d_ws, size_t ws_size,
                              hipStream_t stream) {
  (void)in_sizes; (void)n_in; (void)out_size; (void)ws_size;
  const float* x_in  = (const float*)d_in[0];
  const float* Wq    = (const float*)d_in[1];
  const float* Wk    = (const float*)d_in[2];
  const float* Wv    = (const float*)d_in[3];
  const float* Wo    = (const float*)d_in[4];
  const float* W1    = (const float*)d_in[5];
  const float* b1    = (const float*)d_in[6];
  const float* W2    = (const float*)d_in[7];
  const float* b2    = (const float*)d_in[8];
  const float* gamma = (const float*)d_in[9];
  const float* beta  = (const float*)d_in[10];

  char* p = (char*)d_ws;
  auto alloc = [&](size_t bytes) -> void* {
    void* r = (void*)p;
    p += (bytes + 255) & ~(size_t)255;
    return r;
  };
  const size_t nTok = (size_t)B_ * S_ * D_;
  bf16*  xb   = (bf16*) alloc(nTok * 2);
  float* xc   = (float*)alloc(nTok * 4);
  float* tmp  = (float*)alloc(nTok * 4);
  bf16*  Qb   = (bf16*) alloc((size_t)B_ * H_ * S_ * DK_ * 2);
  bf16*  Kb   = (bf16*) alloc((size_t)B_ * H_ * S_ * DK_ * 2);
  bf16*  Vtb  = (bf16*) alloc((size_t)B_ * H_ * DV_ * S_ * 2);
  bf16*  conc = (bf16*) alloc((size_t)B_ * S_ * H_ * DV_ * 2);
  bf16*  h1b  = (bf16*) alloc((size_t)B_ * S_ * DFF_ * 2);
  bf16*  Wqt  = (bf16*) alloc((size_t)NL_ * H_ * DK_ * D_ * 2);
  bf16*  Wkt  = (bf16*) alloc((size_t)NL_ * H_ * DK_ * D_ * 2);
  bf16*  Wvt  = (bf16*) alloc((size_t)NL_ * H_ * DK_ * D_ * 2);
  bf16*  Wot  = (bf16*) alloc((size_t)NL_ * D_ * H_ * DV_ * 2);
  bf16*  W1t  = (bf16*) alloc((size_t)NL_ * DFF_ * D_ * 2);
  bf16*  W2t  = (bf16*) alloc((size_t)NL_ * D_ * DFF_ * 2);

  dim3 tb(16, 16);
  transpose_cvt_kernel<<<dim3(DK_ / 16, D_ / 16, NL_ * H_), tb, 0, stream>>>(Wq, Wqt, D_, DK_);
  transpose_cvt_kernel<<<dim3(DK_ / 16, D_ / 16, NL_ * H_), tb, 0, stream>>>(Wk, Wkt, D_, DK_);
  transpose_cvt_kernel<<<dim3(DK_ / 16, D_ / 16, NL_ * H_), tb, 0, stream>>>(Wv, Wvt, D_, DK_);
  transpose_cvt_kernel<<<dim3(D_ / 16, (H_ * DV_) / 16, NL_), tb, 0, stream>>>(Wo, Wot, H_ * DV_, D_);
  transpose_cvt_kernel<<<dim3(DFF_ / 16, D_ / 16, NL_), tb, 0, stream>>>(W1, W1t, D_, DFF_);
  transpose_cvt_kernel<<<dim3(D_ / 16, DFF_ / 16, NL_), tb, 0, stream>>>(W2, W2t, DFF_, D_);

  cvt_copy_kernel<<<(int)(nTok / 256), 256, 0, stream>>>(x_in, xc, xb, (int)nTok);

  const int M = B_ * S_;
  for (int l = 0; l < NL_; ++l) {
    qkv_gemm_kernel<false><<<dim3(S_ / 32, 1, B_ * H_), 32, 0, stream>>>(
        xb, Wqt + (size_t)l * H_ * DK_ * D_, Qb);
    qkv_gemm_kernel<false><<<dim3(S_ / 32, 1, B_ * H_), 32, 0, stream>>>(
        xb, Wkt + (size_t)l * H_ * DK_ * D_, Kb);
    qkv_gemm_kernel<true><<<dim3(S_ / 32, 1, B_ * H_), 32, 0, stream>>>(
        xb, Wvt + (size_t)l * H_ * DK_ * D_, Vtb);

    flash_attn_kernel<<<dim3(S_ / 32, B_ * H_), 32, 0, stream>>>(Qb, Kb, Vtb, conc);

    // x + conc @ Wo -> tmp (f32)
    gemm_bf16_kernel<false, false, true, true, false>
        <<<dim3(M / 32, D_ / 32), 32, 0, stream>>>(
            conc, Wot + (size_t)l * D_ * H_ * DV_, nullptr, xc, tmp, nullptr,
            M, D_, H_ * DV_);
    layernorm_kernel<<<M, 256, 0, stream>>>(tmp, gamma + 2 * l, beta + 2 * l, xc, xb);

    // relu(x @ W1 + b1) -> h1b (bf16)
    gemm_bf16_kernel<true, true, false, false, true>
        <<<dim3(M / 32, DFF_ / 32), 32, 0, stream>>>(
            xb, W1t + (size_t)l * DFF_ * D_, b1 + (size_t)l * DFF_, nullptr,
            nullptr, h1b, M, DFF_, D_);
    // x + (h1 @ W2 + b2) -> tmp (f32)
    gemm_bf16_kernel<true, false, true, true, false>
        <<<dim3(M / 32, D_ / 32), 32, 0, stream>>>(
            h1b, W2t + (size_t)l * D_ * DFF_, b2 + (size_t)l * D_, xc, tmp,
            nullptr, M, D_, DFF_);

    float* fOut = (l == NL_ - 1) ? (float*)d_out : xc;
    layernorm_kernel<<<M, 256, 0, stream>>>(tmp, gamma + 2 * l + 1, beta + 2 * l + 1,
                                            fOut, xb);
  }
}
